// APPNPXBN_55121610277361
// MI455X (gfx1250) — compile-verified, matrix operand order checked by reading, not run on hardware
//
#include <hip/hip_runtime.h>

#define HID   64
#define INF   128
#define BN_EPS 1e-5f

typedef __bf16 bf16;
typedef __attribute__((ext_vector_type(16))) __bf16 v16bf;
typedef __attribute__((ext_vector_type(8)))  float  v8f;

// ---------------------------------------------------------------------------
// GEMM: C[M x 64] = A[M x K] * B[K x 64], bf16 WMMA, f32 accumulate.
// 128 threads = 4 waves; each wave owns one 16x16 N-tile of a 16-row block.
// A tile (16xK) row-major and B (Kx64) TRANSPOSED to [n][k] in LDS so every
// lane's fragment is two contiguous 16B runs -> ds_load_b128.
// Requires M % 16 == 0 (true here: 50000 = 3125*16).
// ---------------------------------------------------------------------------
template <int K>
__global__ __launch_bounds__(128)
void gemm_bf16_wmma(const float* __restrict__ A, const float* __restrict__ B,
                    float* __restrict__ C) {
  constexpr int S = K + 4;            // padded row stride (bank-conflict free)
  __shared__ bf16 sBt[HID * S];       // B transposed: [n][k]
  __shared__ bf16 sA[16 * S];         // A tile:       [m][k]
  const int t    = threadIdx.x;
  const int row0 = blockIdx.x * 16;

  for (int i = t; i < K * HID; i += 128) {         // coalesced global read
    int k = i >> 6, n = i & (HID - 1);             // B[k][n]
    sBt[n * S + k] = (bf16)B[i];
  }
  const float* Ab = A + (size_t)row0 * K;
  for (int i = t; i < 16 * K; i += 128) {
    int r = i / K, c = i - r * K;
    sA[r * S + c] = (bf16)Ab[i];
  }
  __syncthreads();

  const int lane = t & 31;
  const int wave = t >> 5;            // N-tile select (0..3)
  const int hlf  = lane >> 4;         // lane half
  const int mn   = lane & 15;         // A row / B col within tile
  const int ncol = wave * 16 + mn;

  const bf16* __restrict__ pa = &sA [mn   * S];
  const bf16* __restrict__ pb = &sBt[ncol * S];

  v8f acc = {};
#pragma unroll
  for (int kk = 0; kk < K; kk += 32) {
    v16bf a, b;
#pragma unroll
    for (int j = 0; j < 8; ++j) {     // two contiguous 8x-bf16 (16B) runs each
      a[j]     = pa[kk + hlf * 8 + j];
      a[j + 8] = pa[kk + 16 + hlf * 8 + j];
      b[j]     = pb[kk + hlf * 8 + j];
      b[j + 8] = pb[kk + 16 + hlf * 8 + j];
    }
    acc = __builtin_amdgcn_wmma_f32_16x16x32_bf16(false, a, false, b,
                                                  (short)0, acc, false, false);
  }

  // C/D layout: VGPR r holds row hlf*8 + r, column = ncol
  float* Cp = C + (size_t)(row0 + hlf * 8) * HID + ncol;
#pragma unroll
  for (int r = 0; r < 8; ++r) Cp[r * HID] = acc[r];
}

// ---------------------------------------------------------------------------
// Graph-norm helpers
// ---------------------------------------------------------------------------
__global__ void fill_k(float* p, float v, int n) {
  int i = blockIdx.x * blockDim.x + threadIdx.x;
  if (i < n) p[i] = v;
}
__global__ void deg_k(const int* __restrict__ dst, float* deg, int E) {
  int e = blockIdx.x * blockDim.x + threadIdx.x;
  if (e < E) atomicAdd(&deg[dst[e]], 1.0f);
}
__global__ void rsqrt_k(float* d, int n) {
  int i = blockIdx.x * blockDim.x + threadIdx.x;
  if (i < n) { float v = d[i]; d[i] = (v > 0.0f) ? rsqrtf(v) : 0.0f; }
}
__global__ void ew_k(const int* __restrict__ src, const int* __restrict__ dst,
                     const float* __restrict__ dinv, float* ew, int E) {
  int e = blockIdx.x * blockDim.x + threadIdx.x;
  if (e < E) ew[e] = dinv[src[e]] * dinv[dst[e]];
}

// out = dinv^2 * in + bias   (self-loop term of the propagation, + layer bias)
__global__ void prop_init_k(const float* __restrict__ in, float* __restrict__ out,
                            const float* __restrict__ dinv,
                            const float* __restrict__ bias, int N) {
  int i = blockIdx.x * blockDim.x + threadIdx.x;
  if (i >= N * HID) return;
  float sw = dinv[i >> 6]; sw *= sw;
  out[i] = sw * in[i] + bias[i & (HID - 1)];
}

// out = (1-alpha) * dinv^2 * in + alpha * h0   (APPNP self-loop + teleport)
__global__ void appnp_init_k(const float* __restrict__ in, const float* __restrict__ h0,
                             float* __restrict__ out, const float* __restrict__ dinv,
                             float alpha, int N) {
  int i = blockIdx.x * blockDim.x + threadIdx.x;
  if (i >= N * HID) return;
  float sw = dinv[i >> 6]; sw *= sw;
  out[i] = (1.0f - alpha) * sw * in[i] + alpha * h0[i];
}

// out[dst] += scale * w * in[src] ; one wave per edge, 2 floats/lane (wave32)
__global__ __launch_bounds__(256)
void edge_scatter_k(const float* __restrict__ in, float* __restrict__ out,
                    const int* __restrict__ src, const int* __restrict__ dst,
                    const float* __restrict__ ew, int E, float scale) {
  int e = blockIdx.x * 8 + (threadIdx.x >> 5);
  if (e >= E) return;
  int lane = threadIdx.x & 31;
  int s = src[e], d = dst[e];
  float w = ew[e] * scale;
  const float* pi = in  + (size_t)s * HID;
  float*       po = out + (size_t)d * HID;
  atomicAdd(&po[lane],      w * pi[lane]);
  atomicAdd(&po[lane + 32], w * pi[lane + 32]);
}

// ---------------------------------------------------------------------------
// BatchNorm (training-mode batch stats) + ReLU
// ---------------------------------------------------------------------------
__global__ void zero_stats_k(float* s) {
  if (threadIdx.x < 2 * HID) s[threadIdx.x] = 0.0f;
}
__global__ __launch_bounds__(256)
void bn_stats_k(const float* __restrict__ h, float* __restrict__ stats, int N) {
  int f  = threadIdx.x & (HID - 1);
  int r0 = blockIdx.x * (256 / HID) + (threadIdx.x >> 6);
  int rs = gridDim.x * (256 / HID);
  float s = 0.0f, s2 = 0.0f;
  for (int r = r0; r < N; r += rs) {
    float v = h[(size_t)r * HID + f];
    s += v; s2 += v * v;
  }
  atomicAdd(&stats[f], s);
  atomicAdd(&stats[HID + f], s2);
}
__global__ void bn_relu_k(float* __restrict__ h, const float* __restrict__ g,
                          const float* __restrict__ b, const float* __restrict__ stats,
                          int N) {
  int i = blockIdx.x * blockDim.x + threadIdx.x;
  if (i >= N * HID) return;
  int f = i & (HID - 1);
  float inv_n = 1.0f / (float)N;
  float m = stats[f] * inv_n;
  float var = stats[HID + f] * inv_n - m * m;
  float y = g[f] * (h[i] - m) * rsqrtf(var + BN_EPS) + b[f];
  h[i] = y > 0.0f ? y : 0.0f;
}

// ---------------------------------------------------------------------------
// Fused bias + log_softmax over 64 classes: one wave per row, 2 cols/lane
// ---------------------------------------------------------------------------
__global__ __launch_bounds__(256)
void lsm_k(const float* __restrict__ z, const float* __restrict__ bfc,
           float* __restrict__ out, int N) {
  int row = blockIdx.x * 8 + (threadIdx.x >> 5);
  if (row >= N) return;
  int lane = threadIdx.x & 31;
  float v0 = z[(size_t)row * HID + lane]      + bfc[lane];
  float v1 = z[(size_t)row * HID + lane + 32] + bfc[lane + 32];
  float mx = fmaxf(v0, v1);
  for (int o = 16; o; o >>= 1) mx = fmaxf(mx, __shfl_xor(mx, o, 32));
  float e = __expf(v0 - mx) + __expf(v1 - mx);
  for (int o = 16; o; o >>= 1) e += __shfl_xor(e, o, 32);
  float lse = mx + __logf(e);
  out[(size_t)row * HID + lane]      = v0 - lse;
  out[(size_t)row * HID + lane + 32] = v1 - lse;
}

// ---------------------------------------------------------------------------
extern "C" void kernel_launch(void* const* d_in, const int* in_sizes, int n_in,
                              void* d_out, int out_size, void* d_ws, size_t ws_size,
                              hipStream_t stream) {
  const float* x   = (const float*)d_in[0];
  const int*   ei  = (const int*)  d_in[1];
  const float* W1  = (const float*)d_in[2];
  const float* b1  = (const float*)d_in[3];
  const float* W2  = (const float*)d_in[4];
  const float* b2  = (const float*)d_in[5];
  const float* Wx  = (const float*)d_in[6];
  const float* bx  = (const float*)d_in[7];
  const float* g1  = (const float*)d_in[8];
  const float* be1 = (const float*)d_in[9];
  const float* g2  = (const float*)d_in[10];
  const float* be2 = (const float*)d_in[11];
  const float* g3  = (const float*)d_in[12];
  const float* be3 = (const float*)d_in[13];
  const float* Wfc = (const float*)d_in[14];
  const float* bfc = (const float*)d_in[15];

  const int N = in_sizes[0] / INF;
  const int E = in_sizes[1] / 2;
  const int* src = ei;
  const int* dst = ei + E;

  float* ws    = (float*)d_ws;
  float* dinv  = ws;  ws += N;
  float* ew    = ws;  ws += E;
  float* h0    = ws;  ws += (size_t)N * HID;
  float* ha    = ws;  ws += (size_t)N * HID;
  float* hb    = ws;  ws += (size_t)N * HID;
  float* stats = ws;  ws += 2 * HID;

  const int TB = 256;
  const int gN   = (N + TB - 1) / TB;
  const int gE   = (E + TB - 1) / TB;
  const int gNH  = (N * HID + TB - 1) / TB;
  const int gEw  = (E + 7) / 8;        // 8 edge-waves per 256-thread block
  const int gRow = (N + 7) / 8;
  const int gMM  = N / 16;             // exact: N % 16 == 0

  // --- GCN normalization: deg (with self-loop), dinv, per-edge weights ---
  fill_k<<<gN, TB, 0, stream>>>(dinv, 1.0f, N);
  deg_k<<<gE, TB, 0, stream>>>(dst, dinv, E);
  rsqrt_k<<<gN, TB, 0, stream>>>(dinv, N);
  ew_k<<<gE, TB, 0, stream>>>(src, dst, dinv, ew, E);

  // --- Conv layers: gemm -> propagate(+bias) -> BN+ReLU ---
  struct { const float *W, *b, *g, *be; } layers[4] = {
    { W1,             b1,       g1, be1 },
    { W2,             b2,       g2, be2 },
    { Wx,             bx,       g3, be3 },
    { Wx + HID * HID, bx + HID, g3, be3 },
  };
  const float* cur_in = x;   // first layer input is x (K=128)
  for (int l = 0; l < 4; ++l) {
    if (l == 0)
      gemm_bf16_wmma<INF><<<gMM, 128, 0, stream>>>(cur_in, layers[l].W, ha);
    else
      gemm_bf16_wmma<HID><<<gMM, 128, 0, stream>>>(cur_in, layers[l].W, ha);
    prop_init_k<<<gNH, TB, 0, stream>>>(ha, hb, dinv, layers[l].b, N);
    edge_scatter_k<<<gEw, TB, 0, stream>>>(ha, hb, src, dst, ew, E, 1.0f);
    zero_stats_k<<<1, 128, 0, stream>>>(stats);
    bn_stats_k<<<256, TB, 0, stream>>>(hb, stats, N);
    bn_relu_k<<<gNH, TB, 0, stream>>>(hb, layers[l].g, layers[l].be, stats, N);
    cur_in = hb;
  }

  // --- APPNP: h0 = h; 10 x [ h = 0.9*prop(h) + 0.1*h0 ] ---
  hipMemcpyAsync(h0, hb, (size_t)N * HID * sizeof(float),
                 hipMemcpyDeviceToDevice, stream);
  float* cur = hb;
  float* nxt = ha;
  for (int k = 0; k < 10; ++k) {
    appnp_init_k<<<gNH, TB, 0, stream>>>(cur, h0, nxt, dinv, 0.1f, N);
    edge_scatter_k<<<gEw, TB, 0, stream>>>(cur, nxt, src, dst, ew, E, 0.9f);
    float* t = cur; cur = nxt; nxt = t;
  }

  // --- FC + log_softmax ---
  gemm_bf16_wmma<HID><<<gMM, 128, 0, stream>>>(cur, Wfc, nxt);
  lsm_k<<<gRow, TB, 0, stream>>>(nxt, bfc, (float*)d_out, N);
}
